// SoftmaxRefMatcher_72567767433908
// MI455X (gfx1250) — compile-verified, matrix operand order checked by reading, not run on hardware
//
#include <hip/hip_runtime.h>

typedef __attribute__((ext_vector_type(16))) __bf16 v16bf;
typedef __attribute__((ext_vector_type(8)))  float  v8f;

#define CC     248          // channels
#define CP     256          // channels padded to K-multiple of 32
#define NN     512          // keypoints per frame
#define WW     256          // image width
#define HWALL  65536        // H*W
#define NWIN   4            // windows (B)
#define PARTS  32           // HW partitions (softmax split)
#define HWPART (HWALL / PARTS)   // 2048 pixels
#define PIX    16           // pixels per WMMA iteration
#define ITERS  (HWPART / PIX)    // 128
#define LOG2E100 144.269504088896340736f   // 100 / ln(2)

static __device__ __forceinline__ unsigned short f32_to_bf16(float f) {
    unsigned int u = __float_as_uint(f);
    u += 0x7fffu + ((u >> 16) & 1u);   // round-to-nearest-even
    return (unsigned short)(u >> 16);
}

// ---------------------------------------------------------------------------
// Kernel 1: per-pixel inverse L2 norm of src frames (desc_dense[::2])
// ---------------------------------------------------------------------------
__global__ __launch_bounds__(256) void invnorm_kernel(
    const float* __restrict__ desc_dense, float* __restrict__ invn)
{
    long t = (long)blockIdx.x * 256 + threadIdx.x;    // 0 .. NWIN*HWALL-1
    int b = (int)(t >> 16);
    int h = (int)(t & (HWALL - 1));
    const float* p = desc_dense + ((long)(2 * b) * CC) * HWALL + h;
    float sum = 0.0f;
    for (int c = 0; c < CC; ++c) {
        float x = p[(long)c * HWALL];
        sum = fmaf(x, x, sum);
    }
    float nrm = sqrtf(sum);
    invn[t] = 1.0f / fmaxf(nrm, 1e-12f);
}

// ---------------------------------------------------------------------------
// Kernel 2: pack tgt keypoint descriptors into CDNA5 A-fragment (16x32 bf16)
// layout: aPack[((b*32 + ntile)*8 + kc)*512 + lane*16 + e]
// ISA 16-bit A 16x32: lane group g, element e -> K = (e/8)*16 + g*8 + (e%8)
// ---------------------------------------------------------------------------
__global__ __launch_bounds__(256) void apack_kernel(
    const float* __restrict__ kp_desc, unsigned short* __restrict__ aPack)
{
    int t = blockIdx.x * 256 + threadIdx.x;   // one lane-fragment each, 32768 total
    int lane  = t & 31;
    int kc    = (t >> 5) & 7;
    int ntile = (t >> 8) & 31;
    int b     = t >> 13;
    int n     = ntile * 16 + (lane & 15);
    int g     = lane >> 4;
    const float* src = kp_desc + ((long)(2 * b + 1) * CC) * NN + n;  // tgt frame 2b+1
    unsigned short* dst = aPack + (long)t * 16;
#pragma unroll
    for (int e = 0; e < 16; ++e) {
        int K = kc * 32 + (e >> 3) * 16 + g * 8 + (e & 7);
        float v = (K < CC) ? src[(long)K * NN] : 0.0f;
        dst[e] = f32_to_bf16(v);
    }
}

// ---------------------------------------------------------------------------
// Kernel 3: fused GEMM (bf16 WMMA) + online softmax + coordinate expectation
// grid: NWIN * 2 nblocks * PARTS = 256 blocks, 512 threads (16 waves)
// wave w owns keypoint tile ntile = nb*16 + w (16 keypoints)
// ---------------------------------------------------------------------------
__global__ __launch_bounds__(512) void match_kernel(
    const float* __restrict__ desc_dense,
    const unsigned short* __restrict__ aPack,
    const float* __restrict__ invn,
    float* __restrict__ partials)
{
    __shared__ unsigned short ldsB[8 * 512];   // 8 k-chunks x (32 lanes x 16 bf16) = 8KB

    int blk  = blockIdx.x;
    int part = blk % PARTS;
    int nb   = (blk / PARTS) & 1;
    int b    = blk / (PARTS * 2);

    int tid  = threadIdx.x;
    int wave = tid >> 5;
    int lane = tid & 31;
    int g    = lane >> 4;      // lane group
    int ln   = lane & 15;      // lane within group (= pixel column)

    // --- load persistent A fragments (16 keypoints x 256 channels, bf16) ---
    int ntile = nb * 16 + wave;
    const v16bf* aBase = (const v16bf*)(aPack + ((long)(b * 32 + ntile) * 8) * 512);
    v16bf afrag[8];
#pragma unroll
    for (int kc = 0; kc < 8; ++kc) afrag[kc] = aBase[kc * 32 + lane];

    // --- online softmax state: 8 keypoint rows per lane ---
    float m[8], s[8], su[8], sv[8];
#pragma unroll
    for (int r = 0; r < 8; ++r) { m[r] = -INFINITY; s[r] = 0.f; su[r] = 0.f; sv[r] = 0.f; }

    const float* srcBase = desc_dense + ((long)(2 * b) * CC) * HWALL;  // src frame 2b
    const float* invBase = invn + (long)b * HWALL;
    int hw0 = part * HWPART;

    for (int it = 0; it < ITERS; ++it) {
        int hbase = hw0 + it * PIX;

        __syncthreads();   // previous iteration's readers done
        {
            // stage B: 256 channels x 16 pixels, scaled by invnorm, bf16,
            // written in B-fragment order (K layout mirrors A layout)
            int c  = tid >> 1;
            int p0 = (tid & 1) * 8;
            float vals[8];
            if (c < CC) {
                const float4* sp = (const float4*)(srcBase + (long)c * HWALL + hbase + p0);
                float4 x0 = sp[0], x1 = sp[1];
                vals[0] = x0.x; vals[1] = x0.y; vals[2] = x0.z; vals[3] = x0.w;
                vals[4] = x1.x; vals[5] = x1.y; vals[6] = x1.z; vals[7] = x1.w;
            } else {
#pragma unroll
                for (int i = 0; i < 8; ++i) vals[i] = 0.0f;
            }
            const float4* ip = (const float4*)(invBase + hbase + p0);
            float4 i0 = ip[0], i1 = ip[1];
            float iv[8] = { i0.x, i0.y, i0.z, i0.w, i1.x, i1.y, i1.z, i1.w };

            int kc = c >> 5;
            int kk = c & 31;
            int gg = (kk >> 3) & 1;
            int e  = (kk >> 4) * 8 + (kk & 7);
#pragma unroll
            for (int i = 0; i < 8; ++i) {
                int p = p0 + i;
                ldsB[kc * 512 + (gg * 16 + p) * 16 + e] = f32_to_bf16(vals[i] * iv[i]);
            }
        }
        __syncthreads();

        // --- 16x16 match tile, K = 256 via 8 bf16 WMMAs ---
        v8f acc = {0.f, 0.f, 0.f, 0.f, 0.f, 0.f, 0.f, 0.f};
#pragma unroll
        for (int kc = 0; kc < 8; ++kc) {
            v16bf bfrag = *(const v16bf*)(&ldsB[kc * 512 + lane * 16]);
            acc = __builtin_amdgcn_wmma_f32_16x16x32_bf16(
                false, afrag[kc], false, bfrag, (short)0, acc, false, false);
        }

        // --- online softmax update (lane = pixel column, rows = keypoints) ---
        int   h  = hbase + ln;
        float uu = (float)(h & (WW - 1));
        float vv = (float)(h >> 8);
#pragma unroll
        for (int r = 0; r < 8; ++r) {
            float t  = acc[r] * LOG2E100;            // logits in log2 domain
            float nm = fmaxf(m[r], t);
            float a  = __builtin_amdgcn_exp2f(m[r] - nm);
            float p  = __builtin_amdgcn_exp2f(t - nm);
            s[r]  = s[r]  * a + p;
            su[r] = su[r] * a + p * uu;
            sv[r] = sv[r] * a + p * vv;
            m[r]  = nm;
        }
    }

    // --- reduce across the 16 pixel-columns (lanes within group) ---
#pragma unroll
    for (int mask = 1; mask < 16; mask <<= 1) {
#pragma unroll
        for (int r = 0; r < 8; ++r) {
            float m2  = __shfl_xor(m[r],  mask, 32);
            float s2  = __shfl_xor(s[r],  mask, 32);
            float su2 = __shfl_xor(su[r], mask, 32);
            float sv2 = __shfl_xor(sv[r], mask, 32);
            float nm  = fmaxf(m[r], m2);
            float a1  = __builtin_amdgcn_exp2f(m[r] - nm);
            float a2  = __builtin_amdgcn_exp2f(m2   - nm);
            s[r]  = s[r]  * a1 + s2  * a2;
            su[r] = su[r] * a1 + su2 * a2;
            sv[r] = sv[r] * a1 + sv2 * a2;
            m[r]  = nm;
        }
    }

    if (ln == 0) {
        // lane group g holds rows M = r + 8g  (C/D layout, ISA 7.12.2)
#pragma unroll
        for (int r = 0; r < 8; ++r) {
            int n = ntile * 16 + g * 8 + r;
            long idx = (((long)b * NN + n) * PARTS + part) * 4;
            float4 rec = { m[r], s[r], su[r], sv[r] };
            *(float4*)(partials + idx) = rec;
        }
    }
}

// ---------------------------------------------------------------------------
// Kernel 4: combine partials -> coords; copy scores; write tgt_ids
// ---------------------------------------------------------------------------
__global__ __launch_bounds__(256) void finalize_kernel(
    const float* __restrict__ partials,
    const float* __restrict__ kp_scores,
    float* __restrict__ out)
{
    int t = blockIdx.x * 256 + threadIdx.x;   // 0 .. NWIN*NN-1
    if (t < NWIN * NN) {
        int b = t / NN, n = t % NN;
        const float* pp = partials + (long)t * PARTS * 4;
        float m = -INFINITY, s = 0.f, su = 0.f, sv = 0.f;
        for (int p = 0; p < PARTS; ++p) {
            float4 rec = *(const float4*)(pp + p * 4);
            float nm = fmaxf(m, rec.x);
            float a1 = __builtin_amdgcn_exp2f(m - nm);
            float a2 = __builtin_amdgcn_exp2f(rec.x - nm);
            s  = s  * a1 + rec.y * a2;
            su = su * a1 + rec.z * a2;
            sv = sv * a1 + rec.w * a2;
            m = nm;
        }
        out[(long)t * 2 + 0] = su / s;                      // pseudo u
        out[(long)t * 2 + 1] = sv / s;                      // pseudo v
        out[NWIN * NN * 2 + t] = kp_scores[(long)(2 * b + 1) * NN + n];
    }
    if (t < NWIN) ((int*)out)[NWIN * NN * 2 + NWIN * NN + t] = 2 * t + 1;  // tgt_ids
}

// ---------------------------------------------------------------------------
extern "C" void kernel_launch(void* const* d_in, const int* in_sizes, int n_in,
                              void* d_out, int out_size, void* d_ws, size_t ws_size,
                              hipStream_t stream)
{
    const float* kp_scores  = (const float*)d_in[0];  // (8,1,512)
    const float* kp_desc    = (const float*)d_in[1];  // (8,248,512)
    // d_in[2] scores_dense unused by reference computation
    const float* desc_dense = (const float*)d_in[3];  // (8,248,256,256)
    float* out = (float*)d_out;

    // workspace layout
    float*          invn     = (float*)d_ws;                                   // 4*65536 f32 = 1MB
    unsigned short* aPack    = (unsigned short*)((char*)d_ws + (size_t)NWIN * HWALL * 4);      // 1MB
    float*          partials = (float*)((char*)aPack + (size_t)NWIN * 32 * 8 * 512 * 2);       // 1MB

    invnorm_kernel<<<(NWIN * HWALL) / 256, 256, 0, stream>>>(desc_dense, invn);
    apack_kernel<<<(NWIN * 32 * 8 * 32) / 256, 256, 0, stream>>>(kp_desc, aPack);
    match_kernel<<<NWIN * 2 * PARTS, 512, 0, stream>>>(desc_dense, aPack, invn, partials);
    finalize_kernel<<<(NWIN * NN + 255) / 256, 256, 0, stream>>>(partials, kp_scores, out);
}
